// MambaSequenceClassificationPositional_23261542875728
// MI455X (gfx1250) — matrix operand order repaired and checked by reading, not compiled, not run on hardware
//
#include <hip/hip_runtime.h>
#include <hip/hip_bf16.h>

// Model dims (fixed by the reference)
#define B_SZ      2
#define L_SZ      1024
#define D_SZ      256
#define NL_SZ     4
#define DIN_SZ    512
#define H_SZ      8
#define P_SZ      64
#define NST_SZ    64
#define CONVD_SZ  640
#define DPROJ_SZ  1160
#define DPROJ_PAD 1216            // next multiple of 64 (wave N-footprint)
#define DCONV_SZ  4
#define ML_SZ     (B_SZ * L_SZ)   // 2048 rows

typedef __bf16 bf16_t;
typedef __attribute__((ext_vector_type(16))) __bf16 v16bf;
typedef __attribute__((ext_vector_type(8)))  float  v8f;

union FragU { v16bf v; uint4 u[2]; };

__device__ __forceinline__ float siluf(float x) { return x / (1.f + __expf(-x)); }
__device__ __forceinline__ float geluf(float x) { return 0.5f * x * (1.f + erff(x * 0.70710678118654752f)); }

// ---------------------------------------------------------------------------
// fp32 -> bf16 weight conversion with row zero-padding:
// out has `layers` blocks of Npad x K; rows >= N are zero.
// ---------------------------------------------------------------------------
__global__ __launch_bounds__(256) void f2bf_pad_kernel(const float* __restrict__ in,
                                                       bf16_t* __restrict__ out,
                                                       int layers, int N, int Npad, int K) {
  long i = (long)blockIdx.x * blockDim.x + threadIdx.x;
  long total = (long)layers * Npad * K;
  if (i >= total) return;
  int  k  = (int)(i % K);
  long rl = i / K;
  int  r  = (int)(rl % Npad);
  int  ly = (int)(rl / Npad);
  out[i] = (r < N) ? (bf16_t)in[((long)ly * N + r) * K + k] : (bf16_t)0.f;
}

// ---------------------------------------------------------------------------
// Embedding + positional: x[b,l,:] = emb[ids[b,l],:] + pos[l,:]  (f32 + bf16)
// ---------------------------------------------------------------------------
__global__ __launch_bounds__(256) void embed_kernel(const int* __restrict__ ids,
                                                    const float* __restrict__ emb,
                                                    const float* __restrict__ pos,
                                                    float* __restrict__ x,
                                                    bf16_t* __restrict__ xbf) {
  int i = blockIdx.x * blockDim.x + threadIdx.x;
  if (i >= ML_SZ * D_SZ) return;
  int d  = i % D_SZ;
  int bl = i / D_SZ;
  int l  = bl % L_SZ;
  float v = emb[ids[bl] * D_SZ + d] + pos[l * D_SZ + d];
  x[i]   = v;
  xbf[i] = (bf16_t)v;
}

// ---------------------------------------------------------------------------
// WMMA bf16 GEMM:  C[M,N] (+= Cres) = A[M,K] * B[Npad,K]^T.
// One wave computes a 16x64 tile: the A fragment is loaded once per k-step
// and reused across 4 v_wmma_f32_16x16x32_bf16 issues (no EXEC divergence in
// the inner loop; B is zero-padded to Npad rows).
// ---------------------------------------------------------------------------
__global__ __launch_bounds__(256) void wmma_gemm_bf16(const bf16_t* __restrict__ A,
                                                      const bf16_t* __restrict__ B,
                                                      float* __restrict__ C,
                                                      int M, int N, int K, int Npad,
                                                      const float* __restrict__ Cres,
                                                      bf16_t* __restrict__ Cbf) {
  int wave = (blockIdx.x * blockDim.x + threadIdx.x) >> 5;
  int lane = threadIdx.x & 31;
  int ngrp = Npad >> 6;                       // 64-column groups
  int mt = wave / ngrp;
  int ng = wave - mt * ngrp;
  if (mt * 16 >= M) return;                   // whole-wave uniform exit
  int m0 = mt * 16, n0 = ng * 64;

  // A fragment: lane holds row m0+(lane&15); K{0..15} contiguous (+16 hi half)
  const bf16_t* Ap = A + (size_t)(m0 + (lane & 15)) * K + ((lane >> 4) ? 16 : 0);
  // B fragment: lane holds weight row (output col) n0+(lane&15); +16 for hi half
  const bf16_t* Bp = B + (size_t)(n0 + (lane & 15)) * K + ((lane >> 4) ? 16 : 0);
  const size_t bstep = (size_t)16 * K;        // next 16-column tile

  v8f acc0 = {0.f, 0.f, 0.f, 0.f, 0.f, 0.f, 0.f, 0.f};
  v8f acc1 = acc0, acc2 = acc0, acc3 = acc0;

  for (int kk = 0; kk < K; kk += 32) {
    FragU a, b0, b1, b2, b3;
    a.u[0]  = *(const uint4*)(Ap + kk);
    a.u[1]  = *(const uint4*)(Ap + kk + 8);
    b0.u[0] = *(const uint4*)(Bp + kk);
    b0.u[1] = *(const uint4*)(Bp + kk + 8);
    b1.u[0] = *(const uint4*)(Bp + bstep + kk);
    b1.u[1] = *(const uint4*)(Bp + bstep + kk + 8);
    b2.u[0] = *(const uint4*)(Bp + 2 * bstep + kk);
    b2.u[1] = *(const uint4*)(Bp + 2 * bstep + kk + 8);
    b3.u[0] = *(const uint4*)(Bp + 3 * bstep + kk);
    b3.u[1] = *(const uint4*)(Bp + 3 * bstep + kk + 8);
    __builtin_prefetch(Ap + kk + 32, 0, 1);   // speculative, fault-free
    __builtin_prefetch(Bp + kk + 32, 0, 1);
    acc0 = __builtin_amdgcn_wmma_f32_16x16x32_bf16(false, a.v, false, b0.v,
                                                   (short)0, acc0, false, false);
    acc1 = __builtin_amdgcn_wmma_f32_16x16x32_bf16(false, a.v, false, b1.v,
                                                   (short)0, acc1, false, false);
    acc2 = __builtin_amdgcn_wmma_f32_16x16x32_bf16(false, a.v, false, b2.v,
                                                   (short)0, acc2, false, false);
    acc3 = __builtin_amdgcn_wmma_f32_16x16x32_bf16(false, a.v, false, b3.v,
                                                   (short)0, acc3, false, false);
  }

  // C layout: VGPR r -> row m0 + r + (lane>=16 ? 8 : 0), col group + (lane&15)
  int cm = m0 + ((lane >> 4) << 3);
  v8f accs[4] = {acc0, acc1, acc2, acc3};
#pragma unroll
  for (int g = 0; g < 4; g++) {
    int cn = n0 + g * 16 + (lane & 15);
    if (cn < N) {
#pragma unroll
      for (int r = 0; r < 8; r++) {
        size_t idx = (size_t)(cm + r) * N + cn;
        float v = accs[g][r];
        if (Cres) v += Cres[idx];
        C[idx] = v;
        if (Cbf) Cbf[idx] = (bf16_t)v;
      }
    }
  }
}

// ---------------------------------------------------------------------------
// Depthwise causal conv (k=4) + bias + SiLU over zxbcdt[:, :, DIN:DIN+640],
// split into xs (512), Bm (64), Cm (64).
// ---------------------------------------------------------------------------
__global__ __launch_bounds__(256) void conv_kernel(const float* __restrict__ zx,
                                                   const float* __restrict__ cw,
                                                   const float* __restrict__ cb,
                                                   float* __restrict__ xs,
                                                   float* __restrict__ Bm,
                                                   float* __restrict__ Cm) {
  int i = blockIdx.x * blockDim.x + threadIdx.x;
  if (i >= ML_SZ * CONVD_SZ) return;
  int c  = i % CONVD_SZ;
  int bl = i / CONVD_SZ;
  int l  = bl % L_SZ;
  int b  = bl / L_SZ;
  float acc = cb[c];
#pragma unroll
  for (int j = 0; j < DCONV_SZ; j++) {
    int ll = l - (DCONV_SZ - 1) + j;
    if (ll >= 0)
      acc += cw[c * DCONV_SZ + j] *
             zx[(size_t)(b * L_SZ + ll) * DPROJ_SZ + DIN_SZ + c];
  }
  float s = siluf(acc);
  if (c < DIN_SZ)               xs[(size_t)bl * DIN_SZ + c] = s;
  else if (c < DIN_SZ + NST_SZ) Bm[(size_t)bl * NST_SZ + (c - DIN_SZ)] = s;
  else                          Cm[(size_t)bl * NST_SZ + (c - DIN_SZ - NST_SZ)] = s;
}

// ---------------------------------------------------------------------------
// dt = softplus(zx[..., -H:] + dt_bias);  dA = exp(dt * (-exp(A_log)))
// ---------------------------------------------------------------------------
__global__ __launch_bounds__(256) void dt_kernel(const float* __restrict__ zx,
                                                 const float* __restrict__ dtb,
                                                 const float* __restrict__ Alog,
                                                 float* __restrict__ dt,
                                                 float* __restrict__ dA) {
  int i = blockIdx.x * blockDim.x + threadIdx.x;
  if (i >= ML_SZ * H_SZ) return;
  int h  = i % H_SZ;
  int bl = i / H_SZ;
  float raw = zx[(size_t)bl * DPROJ_SZ + (DPROJ_SZ - H_SZ) + h] + dtb[h];
  float d = (raw > 20.f) ? raw : log1pf(__expf(raw));
  float A = -__expf(Alog[h]);
  dt[i] = d;
  dA[i] = __expf(d * A);
}

// ---------------------------------------------------------------------------
// Selective scan: one block per (b,h); state h[64p x 64n] in registers,
// 16 n-values per thread; y-dot reduced across the 4 lanes sharing p.
// ---------------------------------------------------------------------------
__global__ __launch_bounds__(256) void scan_kernel(const float* __restrict__ xs,
                                                   const float* __restrict__ Bm,
                                                   const float* __restrict__ Cm,
                                                   const float* __restrict__ dt,
                                                   const float* __restrict__ dA,
                                                   const float* __restrict__ Dp,
                                                   float* __restrict__ y) {
  int bh = blockIdx.x;
  int b  = bh / H_SZ;
  int h  = bh % H_SZ;
  int t  = threadIdx.x;           // 256
  int p  = t >> 2;                // 0..63
  int nb = (t & 3) * 16;          // 16-wide n block
  float st[16];
#pragma unroll
  for (int i = 0; i < 16; i++) st[i] = 0.f;
  float Dph = Dp[h];

  for (int l = 0; l < L_SZ; l++) {
    size_t bl = (size_t)b * L_SZ + l;
    float da = dA[bl * H_SZ + h];
    float d  = dt[bl * H_SZ + h];
    float xp = xs[bl * DIN_SZ + h * P_SZ + p];
    const float* Bl = Bm + bl * NST_SZ + nb;
    const float* Cl = Cm + bl * NST_SZ + nb;
    float dx  = d * xp;
    float acc = 0.f;
#pragma unroll
    for (int i = 0; i < 16; i++) {
      st[i] = da * st[i] + dx * Bl[i];
      acc  += st[i] * Cl[i];
    }
    acc += __shfl_xor(acc, 1, 32);
    acc += __shfl_xor(acc, 2, 32);
    if ((t & 3) == 0)
      y[bl * DIN_SZ + h * P_SZ + p] = acc + Dph * xp;
  }
}

// ---------------------------------------------------------------------------
// Gate with silu(z), RMSNorm, scale by norm_w, emit bf16 rows for out_proj.
// One block per (b,l) row; 256 threads x 2 elements.
// ---------------------------------------------------------------------------
__global__ __launch_bounds__(256) void gatenorm_kernel(const float* __restrict__ y,
                                                       const float* __restrict__ zx,
                                                       const float* __restrict__ nw,
                                                       bf16_t* __restrict__ ybf) {
  __shared__ float red[256];
  int bl = blockIdx.x;
  int t  = threadIdx.x;
  const float* yr = y  + (size_t)bl * DIN_SZ;
  const float* zr = zx + (size_t)bl * DPROJ_SZ;     // z = first DIN channels
  float v0 = yr[t]       * siluf(zr[t]);
  float v1 = yr[t + 256] * siluf(zr[t + 256]);
  red[t] = v0 * v0 + v1 * v1;
  __syncthreads();
  for (int s = 128; s > 0; s >>= 1) {
    if (t < s) red[t] += red[t + s];
    __syncthreads();
  }
  float r = rsqrtf(red[0] * (1.f / DIN_SZ) + 1e-5f);
  ybf[(size_t)bl * DIN_SZ + t]       = (bf16_t)(v0 * r * nw[t]);
  ybf[(size_t)bl * DIN_SZ + t + 256] = (bf16_t)(v1 * r * nw[t + 256]);
}

// ---------------------------------------------------------------------------
// Head: pooled = 0.5*(mean_L + max_L); pooler->gelu; cls1->gelu; cls2.
// One block per batch element; intermediates in LDS.
// ---------------------------------------------------------------------------
__global__ __launch_bounds__(256) void head_kernel(const float* __restrict__ x,
                                                   const float* __restrict__ pw,
                                                   const float* __restrict__ pb,
                                                   const float* __restrict__ c1w,
                                                   const float* __restrict__ c1b,
                                                   const float* __restrict__ c2w,
                                                   const float* __restrict__ c2b,
                                                   float* __restrict__ out) {
  __shared__ float pooled[D_SZ];
  __shared__ float h1[D_SZ];
  __shared__ float h2[D_SZ / 2];
  int b = blockIdx.x;
  int t = threadIdx.x;          // 256 == D_SZ
  float mean = 0.f, mx = -3.402823466e38f;
  for (int l = 0; l < L_SZ; l++) {
    float v = x[((size_t)b * L_SZ + l) * D_SZ + t];
    mean += v;
    mx = fmaxf(mx, v);
  }
  pooled[t] = 0.5f * (mean * (1.f / L_SZ) + mx);
  __syncthreads();
  {
    float a = pb[t];
    for (int d = 0; d < D_SZ; d++) a += pooled[d] * pw[t * D_SZ + d];
    h1[t] = geluf(a);
  }
  __syncthreads();
  if (t < D_SZ / 2) {
    float a = c1b[t];
    for (int d = 0; d < D_SZ; d++) a += h1[d] * c1w[t * D_SZ + d];
    h2[t] = geluf(a);
  }
  __syncthreads();
  if (t < 2) {
    float a = c2b[t];
    for (int d = 0; d < D_SZ / 2; d++) a += h2[d] * c2w[t * (D_SZ / 2) + d];
    out[b * 2 + t] = a;
  }
}

// ---------------------------------------------------------------------------
extern "C" void kernel_launch(void* const* d_in, const int* in_sizes, int n_in,
                              void* d_out, int out_size, void* d_ws, size_t ws_size,
                              hipStream_t stream) {
  (void)in_sizes; (void)n_in; (void)out_size; (void)ws_size;
  const int*   ids   = (const int*)  d_in[0];
  const float* emb   = (const float*)d_in[1];
  const float* pos   = (const float*)d_in[2];
  const float* winW  = (const float*)d_in[3];
  const float* convW = (const float*)d_in[4];
  const float* convB = (const float*)d_in[5];
  const float* dtb   = (const float*)d_in[6];
  const float* Alog  = (const float*)d_in[7];
  const float* Dpar  = (const float*)d_in[8];
  const float* normW = (const float*)d_in[9];
  const float* woutW = (const float*)d_in[10];
  const float* pw    = (const float*)d_in[11];
  const float* pb    = (const float*)d_in[12];
  const float* c1w   = (const float*)d_in[13];
  const float* c1b   = (const float*)d_in[14];
  const float* c2w   = (const float*)d_in[15];
  const float* c2b   = (const float*)d_in[16];
  float* out = (float*)d_out;

  char* ws = (char*)d_ws;
  size_t off = 0;
  auto carve = [&](size_t bytes) -> char* {
    char* p = ws + off;
    off += (bytes + 255) & ~(size_t)255;
    return p;
  };
  float*  x      = (float*) carve(sizeof(float)  * ML_SZ * D_SZ);
  bf16_t* xbf    = (bf16_t*)carve(sizeof(bf16_t) * ML_SZ * D_SZ);
  float*  zx     = (float*) carve(sizeof(float)  * (size_t)ML_SZ * DPROJ_SZ);
  float*  xs     = (float*) carve(sizeof(float)  * ML_SZ * DIN_SZ);
  float*  Bms    = (float*) carve(sizeof(float)  * ML_SZ * NST_SZ);
  float*  Cms    = (float*) carve(sizeof(float)  * ML_SZ * NST_SZ);
  float*  dtbuf  = (float*) carve(sizeof(float)  * ML_SZ * H_SZ);
  float*  dAbuf  = (float*) carve(sizeof(float)  * ML_SZ * H_SZ);
  float*  ybuf   = (float*) carve(sizeof(float)  * ML_SZ * DIN_SZ);
  bf16_t* ybf    = (bf16_t*)carve(sizeof(bf16_t) * ML_SZ * DIN_SZ);
  bf16_t* winbf  = (bf16_t*)carve(sizeof(bf16_t) * (size_t)NL_SZ * DPROJ_PAD * D_SZ);
  bf16_t* woutbf = (bf16_t*)carve(sizeof(bf16_t) * NL_SZ * D_SZ * DIN_SZ);

  {  // in_proj weights: NL x (1160 x 256) -> NL x (1216 x 256) bf16, zero-padded
    long n = (long)NL_SZ * DPROJ_PAD * D_SZ;
    f2bf_pad_kernel<<<(int)((n + 255) / 256), 256, 0, stream>>>(
        winW, winbf, NL_SZ, DPROJ_SZ, DPROJ_PAD, D_SZ);
  }
  {  // out_proj weights: NL x (256 x 512) bf16 (256 already a multiple of 64)
    long n = (long)NL_SZ * D_SZ * DIN_SZ;
    f2bf_pad_kernel<<<(int)((n + 255) / 256), 256, 0, stream>>>(
        woutW, woutbf, NL_SZ, D_SZ, D_SZ, DIN_SZ);
  }

  embed_kernel<<<(ML_SZ * D_SZ + 255) / 256, 256, 0, stream>>>(ids, emb, pos, x, xbf);

  for (int layer = 0; layer < NL_SZ; layer++) {
    {  // in_proj: zx[2048,1160] = x[2048,256] @ Win^T   (16x64 tile per wave)
      int waves = (ML_SZ / 16) * (DPROJ_PAD / 64);
      wmma_gemm_bf16<<<(waves * 32 + 255) / 256, 256, 0, stream>>>(
          xbf, winbf + (size_t)layer * DPROJ_PAD * D_SZ, zx,
          ML_SZ, DPROJ_SZ, D_SZ, DPROJ_PAD, nullptr, nullptr);
    }
    conv_kernel<<<(ML_SZ * CONVD_SZ + 255) / 256, 256, 0, stream>>>(
        zx, convW + layer * CONVD_SZ * DCONV_SZ, convB + layer * CONVD_SZ,
        xs, Bms, Cms);
    dt_kernel<<<(ML_SZ * H_SZ + 255) / 256, 256, 0, stream>>>(
        zx, dtb + layer * H_SZ, Alog + layer * H_SZ, dtbuf, dAbuf);
    scan_kernel<<<B_SZ * H_SZ, 256, 0, stream>>>(
        xs, Bms, Cms, dtbuf, dAbuf, Dpar + layer * H_SZ, ybuf);
    gatenorm_kernel<<<ML_SZ, 256, 0, stream>>>(
        ybuf, zx, normW + layer * DIN_SZ, ybf);
    {  // out_proj + residual: x += ynorm[2048,512] @ Wout^T ; refresh xbf
      int waves = (ML_SZ / 16) * (D_SZ / 64);
      wmma_gemm_bf16<<<(waves * 32 + 255) / 256, 256, 0, stream>>>(
          ybf, woutbf + (size_t)layer * D_SZ * DIN_SZ, x,
          ML_SZ, D_SZ, DIN_SZ, D_SZ, x, xbf);
    }
  }

  head_kernel<<<B_SZ, 256, 0, stream>>>(x, pw, pb, c1w, c1b, c2w, c2b, out);
}
